// SpiralConvBlock_84310208020707
// MI455X (gfx1250) — matrix operand order repaired and checked by reading, not compiled, not run on hardware
//
#include <hip/hip_runtime.h>
#include <hip/hip_bf16.h>
#include <math.h>

// ---------------------------------------------------------------------------
// Problem constants (from reference): B=4, L=4096, D=1024, FF=4096
// ---------------------------------------------------------------------------
#define B_   4
#define L_   4096
#define D_   1024
#define FF_  4096
#define M_   (B_ * L_)          // 16384 rows for the FFN GEMMs
#define NCHUNK 64               // scan chunks
#define LCHUNK 64               // steps per chunk (NCHUNK*LCHUNK == L_)
#define EPS_ 1e-5f

typedef __attribute__((ext_vector_type(16))) __bf16 v16bf;
typedef __attribute__((ext_vector_type(8)))  float  v8f;
typedef __attribute__((ext_vector_type(4)))  unsigned int u32x4;
typedef __attribute__((ext_vector_type(2)))  float  f32x2;

union ABFrag { u32x4 u[2]; v16bf v; };   // 32 bytes = 16 bf16

__device__ __forceinline__ unsigned short f2bf(float f) {
    unsigned int u = __float_as_uint(f);
    u += 0x7fffu + ((u >> 16) & 1u);     // round-to-nearest-even
    return (unsigned short)(u >> 16);
}

__device__ __forceinline__ void load_phazor(const float* __restrict__ pre,
                                            const float* __restrict__ pim,
                                            int d, float& phr, float& phi) {
    float pr = pre[d], pi = pim[d];
    float mag = sqrtf(pr * pr + pi * pi);
    float e = __expf(-mag) / mag;        // unit direction * exp(-|p|)
    phr = pr * e;
    phi = pi * e;
}

// ---------------------------------------------------------------------------
// LayerNorm over D=1024; optional f32 and/or bf16(row-major) outputs.
// One block (256 threads) per row; each thread handles 4 elements.
// ---------------------------------------------------------------------------
__global__ void ln_kernel(const float* __restrict__ x,
                          const float* __restrict__ gamma,
                          const float* __restrict__ beta,
                          float* __restrict__ outf,
                          unsigned short* __restrict__ outb) {
    int row = blockIdx.x;
    int tid = threadIdx.x;
    const float* xr = x + (size_t)row * D_;
    float v[4], s = 0.f, sq = 0.f;
#pragma unroll
    for (int i = 0; i < 4; ++i) {
        v[i] = xr[tid + (i << 8)];
        s  += v[i];
        sq += v[i] * v[i];
    }
    __shared__ float s1[256], s2[256];
    s1[tid] = s; s2[tid] = sq;
    __syncthreads();
    for (int st = 128; st > 0; st >>= 1) {
        if (tid < st) { s1[tid] += s1[tid + st]; s2[tid] += s2[tid + st]; }
        __syncthreads();
    }
    float mu  = s1[0] * (1.0f / (float)D_);
    float var = s2[0] * (1.0f / (float)D_) - mu * mu;
    float inv = rsqrtf(var + EPS_);
#pragma unroll
    for (int i = 0; i < 4; ++i) {
        int d = tid + (i << 8);
        float y = (v[i] - mu) * inv * gamma[d] + beta[d];
        size_t oi = (size_t)row * D_ + d;
        if (outf) outf[oi] = y;
        if (outb) outb[oi] = f2bf(y);
    }
}

// ---------------------------------------------------------------------------
// Spiral conv as a chunked complex scan: c[t] = ph*c[t-1] + q*h[t], c[-1]=last.
// Pass A: per-chunk local scan (zero init), store chunk-final state.
// ---------------------------------------------------------------------------
__global__ void scan_chunk_kernel(const float* __restrict__ h,
                                  const float* __restrict__ pre,
                                  const float* __restrict__ pim,
                                  const float* __restrict__ qre,
                                  const float* __restrict__ qim,
                                  f32x2* __restrict__ fin) {
    int d     = blockIdx.x * 256 + threadIdx.x;
    int chunk = blockIdx.y;
    int b     = blockIdx.z;
    float phr, phi;
    load_phazor(pre, pim, d, phr, phi);
    float qr = qre[d], qi = qim[d];
    float cr = 0.f, ci = 0.f;
    const float* hp = h + ((size_t)b * L_ + (size_t)chunk * LCHUNK) * D_ + d;
    for (int j = 0; j < LCHUNK; ++j) {
        float hv = hp[(size_t)j * D_];
        float nr = phr * cr - phi * ci + qr * hv;
        float ni = phr * ci + phi * cr + qi * hv;
        cr = nr; ci = ni;
    }
    fin[((size_t)b * NCHUNK + chunk) * D_ + d] = f32x2{cr, ci};
}

// Pass B: sequential carry combine across chunks per (b,d). 4096 lanes total.
__global__ void scan_carry_kernel(const f32x2* __restrict__ fin,
                                  const float* __restrict__ pre,
                                  const float* __restrict__ pim,
                                  const float* __restrict__ lcr,
                                  const float* __restrict__ lci,
                                  f32x2* __restrict__ carry) {
    int idx = blockIdx.x * 256 + threadIdx.x;   // 0..B*D-1
    int b = idx >> 10;
    int d = idx & (D_ - 1);
    float phr, phi;
    load_phazor(pre, pim, d, phr, phi);
    // ph^LCHUNK via 6 complex squarings (LCHUNK = 64 = 2^6)
    float ar = phr, ai = phi;
#pragma unroll
    for (int i = 0; i < 6; ++i) {
        float nr = ar * ar - ai * ai;
        float ni = 2.f * ar * ai;
        ar = nr; ai = ni;
    }
    float cr = lcr[d], ci = lci[d];             // c[-1] = last_conv
    for (int c = 0; c < NCHUNK; ++c) {
        size_t o = ((size_t)b * NCHUNK + c) * D_ + d;
        carry[o] = f32x2{cr, ci};
        f32x2 f = fin[o];
        float nr = ar * cr - ai * ci + f.x;
        float ni = ar * ci + ai * cr + f.y;
        cr = nr; ci = ni;
    }
}

// Pass C: re-scan each chunk seeded with its carry; x1 = x + Re(c).
__global__ void scan_apply_kernel(const float* __restrict__ h,
                                  const float* __restrict__ x,
                                  const f32x2* __restrict__ carry,
                                  const float* __restrict__ pre,
                                  const float* __restrict__ pim,
                                  const float* __restrict__ qre,
                                  const float* __restrict__ qim,
                                  float* __restrict__ x1) {
    int d     = blockIdx.x * 256 + threadIdx.x;
    int chunk = blockIdx.y;
    int b     = blockIdx.z;
    float phr, phi;
    load_phazor(pre, pim, d, phr, phi);
    float qr = qre[d], qi = qim[d];
    f32x2 c0 = carry[((size_t)b * NCHUNK + chunk) * D_ + d];
    float cr = c0.x, ci = c0.y;
    size_t base = ((size_t)b * L_ + (size_t)chunk * LCHUNK) * D_ + d;
    for (int j = 0; j < LCHUNK; ++j) {
        size_t o = base + (size_t)j * D_;
        float hv = h[o];
        float nr = phr * cr - phi * ci + qr * hv;
        float ni = phr * ci + phi * cr + qi * hv;
        cr = nr; ci = ni;
        x1[o] = x[o] + cr;
    }
}

// ---------------------------------------------------------------------------
// Pack a row-major f32 [K,N] weight into the WMMA B-matrix bf16 layout:
// flat = ((kb*(N/16)+nb)*32 + lane)*16 + i
//   lane<16 : col = nb*16+lane,    k = kb*32 + {0..7, 16..23}[i]
//   lane>=16: col = nb*16+lane-16, k = kb*32 + {8..15, 24..31}[i]
// ---------------------------------------------------------------------------
__global__ void pack_b_kernel(const float* __restrict__ src,
                              unsigned short* __restrict__ dst,
                              int K, int N) {
    int flat = blockIdx.x * 256 + threadIdx.x;
    int i    = flat & 15;
    int lane = (flat >> 4) & 31;
    int rest = flat >> 9;
    int nt   = N >> 4;
    int nb   = rest % nt;
    int kb   = rest / nt;
    int n = (nb << 4) + (lane & 15);
    int k = (kb << 5) + ((lane & 16) ? 8 : 0) + (i & 7) + ((i & 8) ? 16 : 0);
    dst[flat] = f2bf(src[(size_t)k * N + n]);
}

// ---------------------------------------------------------------------------
// bf16 WMMA GEMM: C[M,N] = A[M,K] @ B[K,N] (+bias, optional SiLU, optional
// residual). A: row-major bf16. B: pre-packed per-lane layout (above).
// Block = 256 threads = 8 waves. Each wave: one 16-row M-tile x four 16-col
// N-tiles -> block covers 128 x 64 of C. K consumed 32 per v_wmma.
// ---------------------------------------------------------------------------
__global__ void gemm_bf16_wmma(const unsigned short* __restrict__ A,
                               const unsigned short* __restrict__ Bp,
                               const float* __restrict__ bias,
                               int M, int N, int K,
                               const float* __restrict__ resid,
                               float* __restrict__ outf,
                               unsigned short* __restrict__ outb,
                               int do_silu) {
    int lane = threadIdx.x & 31;
    int wave = threadIdx.x >> 5;
    int m0 = blockIdx.y * 128 + wave * 16;
    int n0 = blockIdx.x * 64;

    v8f z = {0.f, 0.f, 0.f, 0.f, 0.f, 0.f, 0.f, 0.f};
    v8f acc[4];
#pragma unroll
    for (int t = 0; t < 4; ++t) acc[t] = z;

    // A fragment addressing per the 16-bit A-matrix layout:
    //   lane<16 : row m0+lane,    K-halves {0..7, 16..23}
    //   lane>=16: row m0+lane-16, K-halves {8..15, 24..31}
    int arow  = m0 + (lane & 15);
    int khalf = (lane & 16) ? 8 : 0;
    const unsigned short* Aln = A + (size_t)arow * K + khalf;

    int kblocks = K >> 5;
    int ntiles  = N >> 4;
    int nb0     = n0 >> 4;

    for (int kb = 0; kb < kblocks; ++kb) {
        ABFrag a;
        const u32x4* pa = (const u32x4*)(Aln + ((size_t)kb << 5));
        a.u[0] = pa[0];      // K = khalf      .. khalf+7
        a.u[1] = pa[2];      // K = khalf+16   .. khalf+23
        __builtin_prefetch(Aln + ((size_t)(kb + 1) << 5), 0, 3);
#pragma unroll
        for (int t = 0; t < 4; ++t) {
            ABFrag bm;
            const u32x4* pb =
                (const u32x4*)(Bp + ((size_t)(kb * ntiles + nb0 + t) * 32 + lane) * 16);
            bm.u[0] = pb[0];
            bm.u[1] = pb[1];
            acc[t] = __builtin_amdgcn_wmma_f32_16x16x32_bf16(
                false, a.v, false, bm.v, (short)0, acc[t], false, false);
        }
    }

    // Epilogue. C layout: VGPR r holds M = m0 + r + (lane>=16 ? 8 : 0),
    // N = n0 + t*16 + (lane & 15).
    int ncol = lane & 15;
    int mofs = (lane & 16) ? 8 : 0;
#pragma unroll
    for (int t = 0; t < 4; ++t) {
        int n = n0 + (t << 4) + ncol;
        float bv = bias[n];
#pragma unroll
        for (int r = 0; r < 8; ++r) {
            int m = m0 + mofs + r;
            float v = acc[t][r] + bv;
            if (do_silu) v = v / (1.0f + __expf(-v));   // SiLU
            size_t oi = (size_t)m * N + n;
            if (resid) v += resid[oi];
            if (outf) outf[oi] = v;
            if (outb) outb[oi] = f2bf(v);
        }
    }
}

// ---------------------------------------------------------------------------
// Launcher
// ---------------------------------------------------------------------------
extern "C" void kernel_launch(void* const* d_in, const int* in_sizes, int n_in,
                              void* d_out, int out_size, void* d_ws, size_t ws_size,
                              hipStream_t stream) {
    const float* x   = (const float*)d_in[0];
    const float* g   = (const float*)d_in[1];
    const float* be  = (const float*)d_in[2];
    const float* pre = (const float*)d_in[3];
    const float* pim = (const float*)d_in[4];
    const float* qre = (const float*)d_in[5];
    const float* qim = (const float*)d_in[6];
    const float* lcr = (const float*)d_in[7];
    const float* lci = (const float*)d_in[8];
    const float* w1  = (const float*)d_in[9];
    const float* b1  = (const float*)d_in[10];
    const float* w2  = (const float*)d_in[11];
    const float* b2  = (const float*)d_in[12];
    float* out = (float*)d_out;

    char* ws = (char*)d_ws;
    size_t off = 0;
    float*          h    = (float*)(ws + off);          off += (size_t)M_ * D_ * 4;   // 64 MiB
    float*          x1   = (float*)(ws + off);          off += (size_t)M_ * D_ * 4;   // 64 MiB
    unsigned short* h2b  = (unsigned short*)(ws + off); off += (size_t)M_ * D_ * 2;   // 32 MiB
    unsigned short* actb = (unsigned short*)(ws + off); off += (size_t)M_ * FF_ * 2;  // 128 MiB
    unsigned short* w1p  = (unsigned short*)(ws + off); off += (size_t)D_ * FF_ * 2;  // 8 MiB
    unsigned short* w2p  = (unsigned short*)(ws + off); off += (size_t)FF_ * D_ * 2;  // 8 MiB
    f32x2*          fin  = (f32x2*)(ws + off);          off += (size_t)B_ * NCHUNK * D_ * 8;
    f32x2*          car  = (f32x2*)(ws + off);          off += (size_t)B_ * NCHUNK * D_ * 8;

    // 1) h = LN(x)
    ln_kernel<<<M_, 256, 0, stream>>>(x, g, be, h, nullptr);

    // 2) spiral conv scan: chunk-local -> carry combine -> apply (+ residual)
    dim3 sgrid(D_ / 256, NCHUNK, B_);
    scan_chunk_kernel<<<sgrid, 256, 0, stream>>>(h, pre, pim, qre, qim, fin);
    scan_carry_kernel<<<(B_ * D_) / 256, 256, 0, stream>>>(fin, pre, pim, lcr, lci, car);
    scan_apply_kernel<<<sgrid, 256, 0, stream>>>(h, x, car, pre, pim, qre, qim, x1);

    // 3) h2 = LN(x1) -> bf16 row-major GEMM input
    ln_kernel<<<M_, 256, 0, stream>>>(x1, g, be, nullptr, h2b);

    // 4) pack weights into WMMA B-layout
    pack_b_kernel<<<(D_ * FF_) / 256, 256, 0, stream>>>(w1, w1p, D_, FF_);
    pack_b_kernel<<<(FF_ * D_) / 256, 256, 0, stream>>>(w2, w2p, FF_, D_);

    // 5) act = silu(h2 @ w1 + b1)  [bf16 out]
    gemm_bf16_wmma<<<dim3(FF_ / 64, M_ / 128), 256, 0, stream>>>(
        h2b, w1p, b1, M_, FF_, D_, nullptr, nullptr, actb, 1);

    // 6) out = act @ w2 + b2 + x1  [f32 out]
    gemm_bf16_wmma<<<dim3(D_ / 64, M_ / 128), 256, 0, stream>>>(
        actb, w2p, b2, M_, D_, FF_, x1, out, nullptr, 0);
}
